// TranslatorModel_37924561223887
// MI455X (gfx1250) — matrix operand order repaired and checked by reading, not compiled, not run on hardware
//
#include <hip/hip_runtime.h>

#define DIV_UP(a,b) (((a)+(b)-1)/(b))

typedef __attribute__((ext_vector_type(16))) _Float16 v16h;
typedef __attribute__((ext_vector_type(8)))  float    v8f;
typedef __attribute__((ext_vector_type(4)))  float    v4f;

// ---------------------------------------------------------------------------
// Pack a row-major f32 weight W (N rows, K cols) -- used as B = W^T (K x N) --
// into f16 WMMA-B lane order: tiles of 32(K) x 16(N), tile-major (n-tile outer,
// k-tile inner), within a tile element (lane*16 + h) holds
// B[k0 + ((lane<16)?0:16) + h][n0 + (lane&15)].
// Npad is padded to a multiple of 128 so the GEMM inner loop is branch-free.
// ---------------------------------------------------------------------------
__global__ void pack_weight_f16(const float* __restrict__ W, _Float16* __restrict__ P,
                                int K, int N, int Kpad, int Npad) {
  long idx = (long)blockIdx.x * blockDim.x + threadIdx.x;
  long total = (long)Npad * Kpad;
  if (idx >= total) return;
  int tile   = (int)(idx >> 9);
  int within = (int)(idx & 511);
  int lane = within >> 4;
  int h    = within & 15;
  int Kt = Kpad >> 5;
  int tn = tile / Kt;
  int tk = tile - tn * Kt;
  int n = tn * 16 + (lane & 15);
  int k = tk * 32 + ((lane < 16) ? 0 : 16) + h;
  float v = (n < N && k < K) ? W[(long)n * K + k] : 0.0f;
  P[idx] = (_Float16)v;
}

// ---------------------------------------------------------------------------
// WMMA GEMM: C[M,N] = A[M,K] (f32, row-major, lda) x Bpacked[K,Npad] (f16) + bias.
// Each wave computes a 16x64 strip (4 accumulator tiles, one shared A frag).
// Block = 64 threads = 2 waves. grid = (Npad/128, M/16); Npad % 128 == 0, so
// every wave owns 4 valid (possibly zero-padded) B tiles -> no branches around
// WMMA, EXEC stays all-ones throughout the MAC loop.
// ---------------------------------------------------------------------------
__global__ __launch_bounds__(64) void gemm_wmma(
    const float* __restrict__ A, int lda,
    const _Float16* __restrict__ Bp,
    const float* __restrict__ bias,
    float* __restrict__ C, int ldc,
    int N, int K) {
  const int lane = threadIdx.x & 31;
  const int wave = threadIdx.x >> 5;
  const int row0 = blockIdx.y * 16;
  const int col0 = (blockIdx.x * 2 + wave) * 64;

  const int r = lane & 15;
  const int kbase = (lane < 16) ? 0 : 16;
  const float* ap = A + (long)(row0 + r) * lda + kbase;
  const int Kt = K >> 5;
  const int c16 = col0 >> 4;

  v8f z = {};
  v8f acc[4] = {z, z, z, z};

  for (int kt = 0; kt < Kt; ++kt) {
    // A fragment: per-lane one contiguous run of 16 f32 (k = 16h + s), convert to f16.
    v4f a0 = *(const v4f*)(ap + kt * 32 + 0);
    v4f a1 = *(const v4f*)(ap + kt * 32 + 4);
    v4f a2 = *(const v4f*)(ap + kt * 32 + 8);
    v4f a3 = *(const v4f*)(ap + kt * 32 + 12);
    v16h af;
#pragma unroll
    for (int i = 0; i < 4; ++i) {
      af[i]      = (_Float16)a0[i];
      af[4 + i]  = (_Float16)a1[i];
      af[8 + i]  = (_Float16)a2[i];
      af[12 + i] = (_Float16)a3[i];
    }
    if (kt + 4 < Kt)
      __builtin_prefetch((const void*)(Bp + ((long)c16 * Kt + kt + 4) * 512), 0, 1);
#pragma unroll
    for (int j = 0; j < 4; ++j) {
      v16h bf = *((const v16h*)(Bp + ((long)(c16 + j) * Kt + kt) * 512) + lane);
      acc[j] = __builtin_amdgcn_wmma_f32_16x16x32_f16(
          false, af, false, bf, (short)0, acc[j], false, false);
    }
  }

  const int rowBase = row0 + ((lane >> 4) << 3);
#pragma unroll
  for (int j = 0; j < 4; ++j) {
    int cb = col0 + j * 16 + (lane & 15);
    if (cb < N) {
      float badd = bias ? bias[cb] : 0.0f;
#pragma unroll
      for (int i = 0; i < 8; ++i)
        C[(long)(rowBase + i) * ldc + cb] = acc[j][i] + badd;
    }
  }
}

// ---------------------------------------------------------------------------
// Elementwise kernels
// ---------------------------------------------------------------------------
__device__ __forceinline__ float sigmoidf_(float x) { return 1.0f / (1.0f + __expf(-x)); }

// GRU gate combine. gi/gh are (64,3072), h is (64,1024) updated in place.
// Optional extra outputs y1/y2 at base + b*ld + c.
__global__ void gru_combine(const float* __restrict__ gi, const float* __restrict__ gh,
                            float* __restrict__ h,
                            float* __restrict__ y1, int ld1,
                            float* __restrict__ y2, int ld2) {
  int idx = blockIdx.x * blockDim.x + threadIdx.x;  // 65536 threads
  int b = idx >> 10, c = idx & 1023;
  float ir = gi[b * 3072 + c], iz = gi[b * 3072 + 1024 + c], in = gi[b * 3072 + 2048 + c];
  float hr = gh[b * 3072 + c], hz = gh[b * 3072 + 1024 + c], hn = gh[b * 3072 + 2048 + c];
  float rr = sigmoidf_(ir + hr);
  float zz = sigmoidf_(iz + hz);
  float nn = tanhf(in + rr * hn);
  float out = (1.0f - zz) * nn + zz * h[b * 1024 + c];
  h[b * 1024 + c] = out;
  if (y1) y1[(long)b * ld1 + c] = out;
  if (y2) y2[(long)b * ld2 + c] = out;
}

__global__ void gather_embed(const float* __restrict__ table, const int* __restrict__ tok,
                             float* __restrict__ out, int n_rows, int width, int do_relu) {
  int row = blockIdx.x;
  if (row >= n_rows) return;
  int t = tok[row];
  for (int c = threadIdx.x; c < width; c += blockDim.x) {
    float v = table[(long)t * width + c];
    if (do_relu) v = fmaxf(v, 0.0f);
    out[(long)row * width + c] = v;
  }
}

// bridge_in[b][c] = (c<1024) ? ctx[63][sb][c] : ctx[0][sb][c]
__global__ void build_bridge_in(const float* __restrict__ ctx, const int* __restrict__ sortind,
                                float* __restrict__ bin) {
  int b = blockIdx.x;
  int sb = sortind[b];
  for (int c = threadIdx.x; c < 2048; c += blockDim.x) {
    const float* src = (c < 1024) ? (ctx + (long)(63 * 64 + sb) * 2048)
                                  : (ctx + (long)sb * 2048);
    bin[(long)b * 2048 + c] = src[c];
  }
}

// h0 = st[:, :1024]; h1 = st[:, 1024:]; also seed q_in[:, 1024:] = h1
__global__ void split_state(const float* __restrict__ st, float* __restrict__ h0,
                            float* __restrict__ h1, float* __restrict__ qin) {
  int idx = blockIdx.x * blockDim.x + threadIdx.x;  // 65536
  int b = idx >> 10, c = idx & 1023;
  h0[idx] = st[(long)b * 2048 + c];
  float v = st[(long)b * 2048 + 1024 + c];
  h1[idx] = v;
  qin[(long)b * 2048 + 1024 + c] = v;
}

// attention scores: s[b][j] = dot(ctx_r[b][j][:], q[b][:]) / 32
// ctx_r[b][j][h] = ctx[(j>>1)*64 + sortind[b]][ (j&1)*1024 + h ]
__global__ void att_scores(const float* __restrict__ ctx, const int* __restrict__ sortind,
                           const float* __restrict__ q, float* __restrict__ s) {
  int bj = blockIdx.x;            // 64*128 blocks
  int b = bj >> 7, j = bj & 127;
  int sb = sortind[b];
  const float* row = ctx + ((long)(j >> 1) * 64 + sb) * 2048 + (long)(j & 1) * 1024;
  const float* qr = q + (long)b * 1024;
  float p = 0.0f;
  for (int c = threadIdx.x; c < 1024; c += 128) p += row[c] * qr[c];
  __shared__ float red[128];
  red[threadIdx.x] = p;
  __syncthreads();
  for (int o = 64; o > 0; o >>= 1) {
    if (threadIdx.x < o) red[threadIdx.x] += red[threadIdx.x + o];
    __syncthreads();
  }
  if (threadIdx.x == 0) s[b * 128 + j] = red[0] * 0.03125f;  // 1/sqrt(1024)
}

// softmax over 128 scores, ac = sum_j att[j]*ctx_r[b][j][:]; fan ac out to
// x0in[:, :1024], qin[:, :1024], acbuf; copy teacher row into x0in[:, 1024:1536].
__global__ void att_ac(const float* __restrict__ ctx, const int* __restrict__ sortind,
                       const float* __restrict__ s, const float* __restrict__ teacher_t,
                       float* __restrict__ x0in, float* __restrict__ qin,
                       float* __restrict__ acbuf) {
  __shared__ float att[128];
  __shared__ float red[256];
  int b = blockIdx.x, tid = threadIdx.x;
  int sb = sortind[b];
  float v = (tid < 128) ? s[b * 128 + tid] : -1e30f;
  red[tid] = v;
  __syncthreads();
  for (int o = 128; o > 0; o >>= 1) {
    if (tid < o) red[tid] = fmaxf(red[tid], red[tid + o]);
    __syncthreads();
  }
  float mx = red[0];
  __syncthreads();
  float e = (tid < 128) ? __expf(v - mx) : 0.0f;
  red[tid] = e;
  __syncthreads();
  for (int o = 128; o > 0; o >>= 1) {
    if (tid < o) red[tid] += red[tid + o];
    __syncthreads();
  }
  float inv = 1.0f / red[0];
  __syncthreads();
  if (tid < 128) att[tid] = e * inv;
  __syncthreads();
  for (int c = tid; c < 1024; c += 256) {
    float acc = 0.0f;
    for (int j = 0; j < 128; ++j)
      acc += att[j] * ctx[((long)(j >> 1) * 64 + sb) * 2048 + (long)(j & 1) * 1024 + c];
    x0in[(long)b * 1536 + c] = acc;
    qin[(long)b * 2048 + c] = acc;
    acbuf[(long)b * 1024 + c] = acc;
  }
  for (int c = tid; c < 512; c += 256)
    x0in[(long)b * 1536 + 1024 + c] = teacher_t[(long)b * 512 + c];
}

// in-place row log-softmax over L columns (row stride ld) + optional scalar add
__global__ void log_softmax_rows(float* __restrict__ X, int ld, int L,
                                 const float* __restrict__ add, int addld) {
  int row = blockIdx.x;
  float* x = X + (long)row * ld;
  __shared__ float red[256];
  float mx = -1e30f;
  for (int c = threadIdx.x; c < L; c += 256) mx = fmaxf(mx, x[c]);
  red[threadIdx.x] = mx;
  __syncthreads();
  for (int o = 128; o > 0; o >>= 1) {
    if (threadIdx.x < o) red[threadIdx.x] = fmaxf(red[threadIdx.x], red[threadIdx.x + o]);
    __syncthreads();
  }
  mx = red[0];
  __syncthreads();
  float sm = 0.0f;
  for (int c = threadIdx.x; c < L; c += 256) sm += __expf(x[c] - mx);
  red[threadIdx.x] = sm;
  __syncthreads();
  for (int o = 128; o > 0; o >>= 1) {
    if (threadIdx.x < o) red[threadIdx.x] += red[threadIdx.x + o];
    __syncthreads();
  }
  float lse = mx + __logf(red[0]);
  float a = add ? add[(long)row * addld] : 0.0f;
  for (int c = threadIdx.x; c < L; c += 256) x[c] = x[c] - lse + a;
}

// copy head log-probs cols [0,1600) into output (ld 32000)
__global__ void copy_head(const float* __restrict__ hs, float* __restrict__ out) {
  long idx = (long)blockIdx.x * blockDim.x + threadIdx.x;
  if (idx >= (long)3072 * 1600) return;
  long row = idx / 1600;
  long c = idx - row * 1600;
  out[row * 32000 + c] = hs[row * 1616 + c];
}

// final state / prev_ctx outputs appended after probs
__global__ void write_finals(const float* __restrict__ h0, const float* __restrict__ h1,
                             const float* __restrict__ ac, float* __restrict__ out) {
  int idx = blockIdx.x * blockDim.x + threadIdx.x;  // 65536
  long base = (long)3072 * 32000;
  out[base + idx] = h0[idx];
  out[base + 65536 + idx] = h1[idx];
  out[base + 131072 + idx] = ac[idx];
}

// ---------------------------------------------------------------------------
extern "C" void kernel_launch(void* const* d_in, const int* in_sizes, int n_in,
                              void* d_out, int out_size, void* d_ws, size_t ws_size,
                              hipStream_t stream) {
  (void)in_sizes; (void)n_in; (void)out_size; (void)ws_size;

  const float* enc_embed = (const float*)d_in[0];
  const float* dec_embed = (const float*)d_in[1];
  const float* bridge_w  = (const float*)d_in[2];
  const float* bridge_b  = (const float*)d_in[3];
  const float* ff_w      = (const float*)d_in[4];
  const float* head_w    = (const float*)d_in[5];
  const float* tail0_p   = (const float*)d_in[6];
  const float* tail0_w   = (const float*)d_in[7];
  const float* tail1_p   = (const float*)d_in[8];
  const float* tail1_w   = (const float*)d_in[9];
  const int* sort_ind = (const int*)d_in[34];
  const int* inputs   = (const int*)d_in[35];
  const int* output   = (const int*)d_in[36];
  float* out = (float*)d_out;

  // ---- workspace bump allocator ----
  char* wsb = (char*)d_ws;
  size_t off = 0;
  auto alloc = [&](size_t bytes) -> void* {
    void* p = wsb + off;
    off = (off + bytes + 255) & ~(size_t)255;
    return p;
  };

  // ---- pack all weights to f16 WMMA-B layout (idempotent each call) ----
  // N padded to multiple of 128 (one full block tile) => branch-free GEMM.
  auto pack = [&](const float* W, int N, int K) -> _Float16* {
    int Kp = (K + 31) & ~31, Np = (N + 127) & ~127;
    _Float16* P = (_Float16*)alloc((size_t)Kp * Np * sizeof(_Float16));
    long total = (long)Kp * Np;
    pack_weight_f16<<<dim3((unsigned)DIV_UP(total, 256)), 256, 0, stream>>>(W, P, K, N, Kp, Np);
    return P;
  };

  _Float16* pk_bridge = pack(bridge_w, 2048, 2048);
  _Float16* pk_ff     = pack(ff_w,     1024, 2048);
  _Float16* pk_head   = pack(head_w,   1602, 1024);
  _Float16* pk_t0p    = pack(tail0_p,  256,  1024);
  _Float16* pk_t0w    = pack(tail0_w,  4800, 256);
  _Float16* pk_t1p    = pack(tail1_p,  64,   1024);
  _Float16* pk_t1w    = pack(tail1_w,  25600, 64);

  _Float16* e_wih[2][2]; _Float16* e_whh[2][2];
  const float* e_bih[2][2]; const float* e_bhh[2][2];
  int enc_in[2] = {512, 2048};
  for (int l = 0; l < 2; ++l)
    for (int d = 0; d < 2; ++d) {
      int base = 10 + (l * 2 + d) * 4;
      e_wih[l][d] = pack((const float*)d_in[base + 0], 3072, enc_in[l]);
      e_whh[l][d] = pack((const float*)d_in[base + 1], 3072, 1024);
      e_bih[l][d] = (const float*)d_in[base + 2];
      e_bhh[l][d] = (const float*)d_in[base + 3];
    }

  _Float16* dc_wih[2]; _Float16* dc_whh[2];
  const float* dc_bih[2]; const float* dc_bhh[2];
  int dec_in[2] = {1536, 1024};
  for (int l = 0; l < 2; ++l) {
    int base = 26 + l * 4;
    dc_wih[l] = pack((const float*)d_in[base + 0], 3072, dec_in[l]);
    dc_whh[l] = pack((const float*)d_in[base + 1], 3072, 1024);
    dc_bih[l] = (const float*)d_in[base + 2];
    dc_bhh[l] = (const float*)d_in[base + 3];
  }

  // ---- activation buffers ----
  float* emb     = (float*)alloc((size_t)4096 * 512 * 4);
  float* xmid    = (float*)alloc((size_t)4096 * 2048 * 4);
  float* ctx     = (float*)alloc((size_t)4096 * 2048 * 4);
  float* GI      = (float*)alloc((size_t)4096 * 3072 * 4);
  float* ghbuf   = (float*)alloc((size_t)64 * 3072 * 4);
  float* gibuf   = (float*)alloc((size_t)64 * 3072 * 4);
  float* h0buf   = (float*)alloc((size_t)64 * 1024 * 4);
  float* h1buf   = (float*)alloc((size_t)64 * 1024 * 4);
  float* st      = (float*)alloc((size_t)64 * 2048 * 4);
  float* bridgein= (float*)alloc((size_t)64 * 2048 * 4);
  float* teacher = (float*)alloc((size_t)3072 * 512 * 4);
  float* q_in    = (float*)alloc((size_t)64 * 2048 * 4);
  float* qbuf    = (float*)alloc((size_t)64 * 1024 * 4);
  float* sbuf    = (float*)alloc((size_t)64 * 128 * 4);
  float* x0in    = (float*)alloc((size_t)64 * 1536 * 4);
  float* acbuf   = (float*)alloc((size_t)64 * 1024 * 4);
  float* flatbuf = (float*)alloc((size_t)3072 * 1024 * 4);
  float* headsc  = (float*)alloc((size_t)3072 * 1616 * 4);
  float* t0h     = (float*)alloc((size_t)3072 * 256 * 4);
  float* t1h     = (float*)alloc((size_t)3072 * 64 * 4);

  auto gemm = [&](const float* A, int lda, const _Float16* Bp, const float* bias,
                  float* C, int ldc, int M, int N, int K) {
    int Np = (N + 127) & ~127;                 // must match pack()
    dim3 grid((unsigned)(Np / 128), (unsigned)(M / 16));
    gemm_wmma<<<grid, 64, 0, stream>>>(A, lda, Bp, bias, C, ldc, N, K);
  };

  // ---- embeddings ----
  gather_embed<<<4096, 128, 0, stream>>>(enc_embed, inputs, emb, 4096, 512, 0);
  gather_embed<<<3072, 128, 0, stream>>>(dec_embed, output, teacher, 3072, 512, 1);

  // ---- encoder: 2 layers x 2 directions ----
  for (int l = 0; l < 2; ++l) {
    const float* xin = (l == 0) ? emb : xmid;
    int lda = enc_in[l];
    float* xout = (l == 0) ? xmid : ctx;
    for (int d = 0; d < 2; ++d) {
      // gi for all timesteps at once (big GEMM off the critical path)
      gemm(xin, lda, e_wih[l][d], e_bih[l][d], GI, 3072, 4096, 3072, lda);
      hipMemsetAsync(h0buf, 0, (size_t)64 * 1024 * 4, stream);
      for (int s = 0; s < 64; ++s) {
        int ss = d ? (63 - s) : s;
        gemm(h0buf, 1024, e_whh[l][d], e_bhh[l][d], ghbuf, 3072, 64, 3072, 1024);
        gru_combine<<<256, 256, 0, stream>>>(GI + (long)ss * 64 * 3072, ghbuf, h0buf,
                                             xout + (long)ss * 64 * 2048 + d * 1024, 2048,
                                             (float*)nullptr, 0);
      }
    }
  }

  // ---- bridge -> initial decoder states ----
  build_bridge_in<<<64, 256, 0, stream>>>(ctx, sort_ind, bridgein);
  gemm(bridgein, 2048, pk_bridge, bridge_b, st, 2048, 64, 2048, 2048);
  hipMemsetAsync(q_in, 0, (size_t)64 * 2048 * 4, stream);  // pc part = 0
  split_state<<<256, 256, 0, stream>>>(st, h0buf, h1buf, q_in);

  // ---- decoder with attention ----
  for (int t = 0; t < 48; ++t) {
    gemm(q_in, 2048, pk_ff, nullptr, qbuf, 1024, 64, 1024, 2048);
    att_scores<<<64 * 128, 128, 0, stream>>>(ctx, sort_ind, qbuf, sbuf);
    att_ac<<<64, 256, 0, stream>>>(ctx, sort_ind, sbuf,
                                   teacher + (long)t * 64 * 512, x0in, q_in, acbuf);
    gemm(x0in, 1536, dc_wih[0], dc_bih[0], gibuf, 3072, 64, 3072, 1536);
    gemm(h0buf, 1024, dc_whh[0], dc_bhh[0], ghbuf, 3072, 64, 3072, 1024);
    gru_combine<<<256, 256, 0, stream>>>(gibuf, ghbuf, h0buf,
                                         (float*)nullptr, 0, (float*)nullptr, 0);
    gemm(h0buf, 1024, dc_wih[1], dc_bih[1], gibuf, 3072, 64, 3072, 1024);
    gemm(h1buf, 1024, dc_whh[1], dc_bhh[1], ghbuf, 3072, 64, 3072, 1024);
    // h1 -> q_in[:,1024:] for next step, and into flat result at row b*48+t
    gru_combine<<<256, 256, 0, stream>>>(gibuf, ghbuf, h1buf,
                                         q_in + 1024, 2048,
                                         flatbuf + (long)t * 1024, 48 * 1024);
  }

  // ---- adaptive log-softmax output ----
  gemm(flatbuf, 1024, pk_head, nullptr, headsc, 1616, 3072, 1602, 1024);
  log_softmax_rows<<<3072, 256, 0, stream>>>(headsc, 1616, 1602, (const float*)nullptr, 0);
  copy_head<<<DIV_UP(3072 * 1600, 256), 256, 0, stream>>>(headsc, out);

  gemm(flatbuf, 1024, pk_t0p, nullptr, t0h, 256, 3072, 256, 1024);
  gemm(t0h, 256, pk_t0w, nullptr, out + 1600, 32000, 3072, 4800, 256);
  log_softmax_rows<<<3072, 256, 0, stream>>>(out + 1600, 32000, 4800, headsc + 1600, 1616);

  gemm(flatbuf, 1024, pk_t1p, nullptr, t1h, 64, 3072, 64, 1024);
  gemm(t1h, 64, pk_t1w, nullptr, out + 6400, 32000, 3072, 25600, 64);
  log_softmax_rows<<<3072, 256, 0, stream>>>(out + 6400, 32000, 25600, headsc + 1601, 1616);

  write_finals<<<256, 256, 0, stream>>>(h0buf, h1buf, acbuf, out);
}
